// DustRiskFormer_31370441130106
// MI455X (gfx1250) — matrix operand order repaired and checked by reading, not compiled, not run on hardware
//
#include <hip/hip_runtime.h>

// ---------------------------------------------------------------------------
// DustRiskFormer for MI455X (gfx1250, wave32, WMMA).
// All dense GEMMs run on v_wmma_f32_16x16x32_f16 with fp32 accumulation.
// Operands are staged as f16 with K padded to multiples of 32 so fragment
// loads are unpredicated global_load_b128 pairs; the K-loop is register
// double-buffered so next-step loads overlap current-step WMMAs.
// ---------------------------------------------------------------------------

typedef __attribute__((ext_vector_type(16))) _Float16 v16h;
typedef __attribute__((ext_vector_type(8)))  _Float16 v8h;
typedef __attribute__((ext_vector_type(8)))  float    v8f;

// Problem constants
#define BB 8
#define TT 48
#define NN 512
#define FF 16
#define SS 12
#define HH 128
#define NHH 8
#define MTOT (BB * NN * TT)     // 196608 temporal rows
#define MG   (BB * NN)          // 4096 graph rows

// ---------------------------------------------------------------------------
// WMMA fragment loaders (CDNA5 ISA 7.12.2, wave32). K must be 32-aligned,
// base pointers 16B-aligned, lda/ldw multiples of 8 halves.
// ---------------------------------------------------------------------------

// A: 16x32 f16 tile at (row0,k0) of row-major A[M][lda].
__device__ __forceinline__ v16h frag_a(const _Float16* __restrict__ A, int lda,
                                       int row0, int k0) {
  const int lane = threadIdx.x & 31;
  const _Float16* p = A + (size_t)(row0 + (lane & 15)) * lda + k0 + ((lane >> 4) << 3);
  v8h lo = *(const v8h*)p;
  v8h hi = *(const v8h*)(p + 16);
  return __builtin_shufflevector(lo, hi, 0, 1, 2, 3, 4, 5, 6, 7,
                                 8, 9, 10, 11, 12, 13, 14, 15);
}

// B: 32x16 tile (k0,col0) from f16 weight W[N][ldw] row-major (A @ W^T).
__device__ __forceinline__ v16h frag_b(const _Float16* __restrict__ W, int ldw,
                                       int k0, int col0) {
  const int lane = threadIdx.x & 31;
  const _Float16* p = W + (size_t)(col0 + (lane & 15)) * ldw + k0 + ((lane >> 4) << 3);
  v8h lo = *(const v8h*)p;
  v8h hi = *(const v8h*)(p + 16);
  return __builtin_shufflevector(lo, hi, 0, 1, 2, 3, 4, 5, 6, 7,
                                 8, 9, 10, 11, 12, 13, 14, 15);
}

// ---------------------------------------------------------------------------
// Generic WMMA GEMM: each wave owns one 16x64 strip (4 N-tiles, A reuse x4),
// double-buffered over K. Requires M%16==0, N%64==0, K%32==0.
// act: 0 none, 1 exact GELU, 2 ReLU.
// ---------------------------------------------------------------------------
__global__ __launch_bounds__(256) void gemm_wmma_kernel(
    const _Float16* __restrict__ A, int lda,
    const _Float16* __restrict__ W, int ldw,
    const float* __restrict__ bias,
    float* __restrict__ C32, int ldc32,
    _Float16* __restrict__ C16, int ldc16,
    int M, int N, int K, int act) {
  const int wave = blockIdx.x * (blockDim.x >> 5) + (threadIdx.x >> 5);
  const int ncols = N >> 6;
  const int nstrips = (M >> 4) * ncols;
  if (wave >= nstrips) return;          // wave-uniform: EXEC all-1s for WMMA
  const int row0 = (wave / ncols) << 4;
  const int nOff = (wave % ncols) << 6;

  v8f acc[4] = {v8f{}, v8f{}, v8f{}, v8f{}};
  // software pipeline: fragments for step k0 live while step k0+32 loads issue
  v16h a  = frag_a(A, lda, row0, 0);
  v16h b0 = frag_b(W, ldw, 0, nOff);
  v16h b1 = frag_b(W, ldw, 0, nOff + 16);
  v16h b2 = frag_b(W, ldw, 0, nOff + 32);
  v16h b3 = frag_b(W, ldw, 0, nOff + 48);
  for (int k0 = 0; k0 < K; k0 += 32) {
    v16h an = a, bn0 = b0, bn1 = b1, bn2 = b2, bn3 = b3;
    const int kn = k0 + 32;
    if (kn < K) {                        // wave-uniform prefetch of next k-step
      an  = frag_a(A, lda, row0, kn);
      bn0 = frag_b(W, ldw, kn, nOff);
      bn1 = frag_b(W, ldw, kn, nOff + 16);
      bn2 = frag_b(W, ldw, kn, nOff + 32);
      bn3 = frag_b(W, ldw, kn, nOff + 48);
    }
    acc[0] = __builtin_amdgcn_wmma_f32_16x16x32_f16(false, a, false, b0, (short)0, acc[0], false, false);
    acc[1] = __builtin_amdgcn_wmma_f32_16x16x32_f16(false, a, false, b1, (short)0, acc[1], false, false);
    acc[2] = __builtin_amdgcn_wmma_f32_16x16x32_f16(false, a, false, b2, (short)0, acc[2], false, false);
    acc[3] = __builtin_amdgcn_wmma_f32_16x16x32_f16(false, a, false, b3, (short)0, acc[3], false, false);
    a = an; b0 = bn0; b1 = bn1; b2 = bn2; b3 = bn3;
  }

  const int lane = threadIdx.x & 31;
  const int nloc = lane & 15;
  const int mbase = row0 + ((lane >> 4) << 3);
  float res[4][8];
#pragma unroll
  for (int t = 0; t < 4; ++t) {
    const int n = nOff + (t << 4) + nloc;
    const float bs = bias ? bias[n] : 0.f;
#pragma unroll
    for (int r = 0; r < 8; ++r) {
      float v = acc[t][r] + bs;
      if (act == 1)      v = 0.5f * v * (1.f + erff(v * 0.70710678118f));
      else if (act == 2) v = fmaxf(v, 0.f);
      res[t][r] = v;
    }
  }
  if (C32) {
#pragma unroll
    for (int t = 0; t < 4; ++t) {
      const int n = nOff + (t << 4) + nloc;
#pragma unroll
      for (int r = 0; r < 8; ++r)
        C32[(size_t)(mbase + r) * ldc32 + n] = res[t][r];
    }
  }
  if (C16) {
#pragma unroll
    for (int t = 0; t < 4; ++t) {
      const int n = nOff + (t << 4) + nloc;
#pragma unroll
      for (int r = 0; r < 8; ++r)
        C16[(size_t)(mbase + r) * ldc16 + n] = (_Float16)res[t][r];
    }
  }
}

// ---------------------------------------------------------------------------
// Staging kernels
// ---------------------------------------------------------------------------

// x (B,T,N,F) -> xr f16 (MTOT, 32) zero-padded:  row = (b*N+n)*T + t
__global__ void xr_stage_kernel(const float* __restrict__ x, _Float16* __restrict__ xr) {
  int idx = blockIdx.x * blockDim.x + threadIdx.x;
  if (idx >= MTOT * 32) return;
  int row = idx >> 5, f = idx & 31;
  int t = row % TT, seq = row / TT;
  int b = seq >> 9, n = seq & 511;
  float v = (f < FF) ? x[(((size_t)b * TT + t) * NN + n) * FF + f] : 0.f;
  xr[idx] = (_Float16)v;
}

// x_static (N,S) -> f16 (N,32) zero-padded
__global__ void xs_stage_kernel(const float* __restrict__ xs, _Float16* __restrict__ o) {
  int idx = blockIdx.x * blockDim.x + threadIdx.x;
  if (idx >= NN * 32) return;
  int r = idx >> 5, c = idx & 31;
  o[idx] = (_Float16)((c < SS) ? xs[r * SS + c] : 0.f);
}

// generic f32 weight [rows][scols] -> f16 [rows][dcols], zero-pad cols
__global__ void w_stage_kernel(const float* __restrict__ src, _Float16* __restrict__ dst,
                               int rows, int scols, int dcols) {
  int idx = blockIdx.x * blockDim.x + threadIdx.x;
  if (idx >= rows * dcols) return;
  int r = idx / dcols, c = idx - r * dcols;
  dst[idx] = (_Float16)((c < scols) ? src[(size_t)r * scols + c] : 0.f);
}

// gh f32 (b*512+j, 128) -> ghT f16 (b, 128, 512): ghT[b][h][j] = gh[b][j][h]
__global__ void ght_kernel(const float* __restrict__ gh, _Float16* __restrict__ ght) {
  int idx = blockIdx.x * blockDim.x + threadIdx.x;
  if (idx >= MG * HH) return;
  int b = idx >> 16, rem = idx & 65535;
  int j = rem >> 7, h = rem & 127;
  ght[((size_t)b * HH + h) * NN + j] = (_Float16)gh[(size_t)idx];
}

// ---------------------------------------------------------------------------
// Fused residual + LayerNorm (one wave per 128-wide row), wave32 reductions
// ---------------------------------------------------------------------------
__global__ __launch_bounds__(256) void add_ln_kernel(
    const float* __restrict__ A, const float* __restrict__ Bv,
    const float* __restrict__ g, const float* __restrict__ be,
    float* __restrict__ out32, _Float16* __restrict__ out16, int M) {
  int row = blockIdx.x * (blockDim.x >> 5) + (threadIdx.x >> 5);
  if (row >= M) return;
  int lane = threadIdx.x & 31;
  const float* pa = A + (size_t)row * HH;
  const float* pb = Bv + (size_t)row * HH;
  float v[4], s = 0.f;
#pragma unroll
  for (int i = 0; i < 4; ++i) { v[i] = pa[lane + 32 * i] + pb[lane + 32 * i]; s += v[i]; }
#pragma unroll
  for (int o = 16; o >= 1; o >>= 1) s += __shfl_xor(s, o);
  float mean = s * (1.f / HH), q = 0.f;
#pragma unroll
  for (int i = 0; i < 4; ++i) { float d = v[i] - mean; q += d * d; }
#pragma unroll
  for (int o = 16; o >= 1; o >>= 1) q += __shfl_xor(q, o);
  float rstd = rsqrtf(q * (1.f / HH) + 1e-5f);
#pragma unroll
  for (int i = 0; i < 4; ++i) {
    int c = lane + 32 * i;
    float y = (v[i] - mean) * rstd * g[c] + be[c];
    if (out32) out32[(size_t)row * HH + c] = y;
    if (out16) out16[(size_t)row * HH + c] = (_Float16)y;
  }
}

// ---------------------------------------------------------------------------
// Multi-head attention (T=48, hd=16): one block per sequence, 4 waves x 2 heads.
// ---------------------------------------------------------------------------
__global__ __launch_bounds__(128) void attn_kernel(
    const _Float16* __restrict__ qkv, _Float16* __restrict__ ao,
    float* __restrict__ t_attn) {
  __shared__ float sc[4][TT * TT];
  __shared__ float tsum[TT * TT];
  const int seq = blockIdx.x;
  const int wv = threadIdx.x >> 5, lane = threadIdx.x & 31;
  for (int i = threadIdx.x; i < TT * TT; i += blockDim.x) tsum[i] = 0.f;
  __syncthreads();

  const _Float16* base = qkv + (size_t)seq * TT * (3 * HH);
  float* s = sc[wv];
  for (int pass = 0; pass < 2; ++pass) {
    const int h = wv + pass * 4;
    const _Float16* q  = base + h * 16;
    const _Float16* k  = base + HH + h * 16;
    const _Float16* vv = base + 2 * HH + h * 16;
    for (int p = lane; p < TT * TT; p += 32) {
      int i = p / TT, j = p - i * TT;
      float acc = 0.f;
#pragma unroll
      for (int d = 0; d < 16; ++d)
        acc += (float)q[i * 384 + d] * (float)k[j * 384 + d];
      s[p] = acc * 0.25f;                     // 1/sqrt(16)
    }
    for (int i = lane; i < TT; i += 32) {     // row softmax (wave-private)
      float mx = -1e30f;
      for (int j = 0; j < TT; ++j) mx = fmaxf(mx, s[i * TT + j]);
      float sum = 0.f;
      for (int j = 0; j < TT; ++j) { float e = __expf(s[i * TT + j] - mx); s[i * TT + j] = e; sum += e; }
      float inv = 1.f / sum;
      for (int j = 0; j < TT; ++j) s[i * TT + j] *= inv;
    }
    for (int p = lane; p < TT * TT; p += 32) atomicAdd(&tsum[p], s[p]);
    for (int p = lane; p < TT * 16; p += 32) {
      int i = p >> 4, d = p & 15;
      float acc = 0.f;
      for (int j = 0; j < TT; ++j) acc += s[i * TT + j] * (float)vv[j * 384 + d];
      ao[((size_t)seq * TT + i) * HH + h * 16 + d] = (_Float16)acc;
    }
  }
  __syncthreads();
  for (int i = threadIdx.x; i < TT * TT; i += blockDim.x)
    t_attn[(size_t)seq * TT * TT + i] = tsum[i] * (1.f / NHH);
}

// ---------------------------------------------------------------------------
// GAT pieces
// ---------------------------------------------------------------------------

// CAT[4096][384] cols 0..255 = [t_repr | s_repr]
__global__ void node_build_kernel(const float* __restrict__ ln2,
                                  const float* __restrict__ srepr,
                                  _Float16* __restrict__ cat) {
  int idx = blockIdx.x * blockDim.x + threadIdx.x;
  if (idx >= MG * 256) return;
  int row = idx >> 8, c = idx & 255;
  float v = (c < HH) ? ln2[((size_t)row * TT + (TT - 1)) * HH + c]
                     : srepr[(size_t)(row & 511) * HH + (c - HH)];
  cat[(size_t)row * 384 + c] = (_Float16)v;
}

__global__ __launch_bounds__(256) void s12_kernel(const float* __restrict__ gh,
                                                  const float* __restrict__ ga,
                                                  float* __restrict__ s1,
                                                  float* __restrict__ s2) {
  int row = blockIdx.x * 8 + (threadIdx.x >> 5);
  if (row >= MG) return;
  int lane = threadIdx.x & 31;
  const float* g = gh + (size_t)row * HH;
  float a = 0.f, b = 0.f;
#pragma unroll
  for (int i = 0; i < 4; ++i) {
    int c = lane + 32 * i;
    a += g[c] * ga[c];
    b += g[c] * ga[HH + c];
  }
#pragma unroll
  for (int o = 16; o >= 1; o >>= 1) { a += __shfl_xor(a, o); b += __shfl_xor(b, o); }
  if (lane == 0) { s1[row] = a; s2[row] = b; }
}

__global__ __launch_bounds__(256) void alpha_kernel(
    const float* __restrict__ s1, const float* __restrict__ s2,
    const int* __restrict__ adj, float* __restrict__ alpha32,
    _Float16* __restrict__ alpha16) {
  int row = blockIdx.x * 8 + (threadIdx.x >> 5);
  if (row >= MG) return;
  int lane = threadIdx.x & 31;
  int b = row >> 9, i = row & 511;
  float si = s1[row];
  const float* s2b = s2 + ((size_t)b << 9);
  const int* adjr = adj + (size_t)i * NN;
  float e[16], mx = -1e30f;
#pragma unroll
  for (int t = 0; t < 16; ++t) {
    int j = lane + 32 * t;
    float v = si + s2b[j];
    v = (v > 0.f) ? v : 0.2f * v;              // leaky_relu(0.2)
    v = (adjr[j] > 0) ? v : -1e30f;            // adjacency mask
    e[t] = v; mx = fmaxf(mx, v);
  }
#pragma unroll
  for (int o = 16; o >= 1; o >>= 1) mx = fmaxf(mx, __shfl_xor(mx, o));
  float sum = 0.f;
#pragma unroll
  for (int t = 0; t < 16; ++t) {
    float ex = (e[t] <= -1e29f) ? 0.f : __expf(e[t] - mx);
    e[t] = ex; sum += ex;
  }
#pragma unroll
  for (int o = 16; o >= 1; o >>= 1) sum += __shfl_xor(sum, o);
  float inv = 1.f / sum;
#pragma unroll
  for (int t = 0; t < 16; ++t) {
    int j = lane + 32 * t;
    float a = e[t] * inv;
    alpha32[(size_t)row * NN + j] = a;
    alpha16[(size_t)row * NN + j] = (_Float16)a;
  }
}

__global__ void heads_kernel(const float* __restrict__ fused,
                             const float* __restrict__ reg_w, const float* __restrict__ reg_b,
                             const float* __restrict__ risk_w, const float* __restrict__ risk_b,
                             const float* __restrict__ warn_w, const float* __restrict__ warn_b,
                             float* __restrict__ reg_o, float* __restrict__ risk_o,
                             float* __restrict__ warn_o) {
  int idx = blockIdx.x * blockDim.x + threadIdx.x;
  int row = idx / 30, o = idx % 30;
  if (row >= MG) return;
  const float* f = fused + (size_t)row * HH;
  const float* w; float bias; float* dst;
  if (o < 6)       { w = reg_w + o * HH; bias = reg_b[o]; dst = reg_o + (size_t)row * 6 + o; }
  else if (o < 24) { int r = o - 6;  w = risk_w + r * HH; bias = risk_b[r]; dst = risk_o + (size_t)row * 18 + r; }
  else             { int r = o - 24; w = warn_w + r * HH; bias = warn_b[r]; dst = warn_o + (size_t)row * 6 + r; }
  float acc = bias;
#pragma unroll 8
  for (int k = 0; k < HH; ++k) acc += f[k] * w[k];
  *dst = acc;
}

// ---------------------------------------------------------------------------
// Host orchestration
// ---------------------------------------------------------------------------
static void launch_gemm(const _Float16* A, int lda, const _Float16* W, int ldw,
                        const float* bias, float* C32, int ldc32,
                        _Float16* C16, int ldc16,
                        int M, int N, int K, int act, hipStream_t s) {
  int strips = (M >> 4) * (N >> 6);
  int blocks = (strips + 7) / 8;
  gemm_wmma_kernel<<<blocks, 256, 0, s>>>(A, lda, W, ldw, bias,
                                          C32, ldc32, C16, ldc16, M, N, K, act);
}

static void stage_w(const float* src, _Float16* dst, int rows, int scols, int dcols,
                    hipStream_t s) {
  int n = rows * dcols;
  w_stage_kernel<<<(n + 255) / 256, 256, 0, s>>>(src, dst, rows, scols, dcols);
}

extern "C" void kernel_launch(void* const* d_in, const int* in_sizes, int n_in,
                              void* d_out, int out_size, void* d_ws, size_t ws_size,
                              hipStream_t stream) {
  (void)in_sizes; (void)n_in; (void)out_size; (void)ws_size;
  const float* x        = (const float*)d_in[0];
  const float* x_static = (const float*)d_in[1];
  const int*   adj      = (const int*)d_in[2];
  const float* proj_w = (const float*)d_in[3];  const float* proj_b = (const float*)d_in[4];
  const float* in_w   = (const float*)d_in[5];  const float* in_b   = (const float*)d_in[6];
  const float* out_w  = (const float*)d_in[7];  const float* out_b  = (const float*)d_in[8];
  const float* ln1_g  = (const float*)d_in[9];  const float* ln1_b  = (const float*)d_in[10];
  const float* ffn_w1 = (const float*)d_in[11]; const float* ffn_b1 = (const float*)d_in[12];
  const float* ffn_w2 = (const float*)d_in[13]; const float* ffn_b2 = (const float*)d_in[14];
  const float* ln2_g  = (const float*)d_in[15]; const float* ln2_b  = (const float*)d_in[16];
  const float* stat_w = (const float*)d_in[17]; const float* stat_b = (const float*)d_in[18];
  const float* gat_w  = (const float*)d_in[19]; const float* gat_a  = (const float*)d_in[20];
  const float* fuse_w = (const float*)d_in[21]; const float* fuse_b = (const float*)d_in[22];
  const float* reg_w  = (const float*)d_in[23]; const float* reg_b  = (const float*)d_in[24];
  const float* risk_w = (const float*)d_in[25]; const float* risk_b = (const float*)d_in[26];
  const float* warn_w = (const float*)d_in[27]; const float* warn_b = (const float*)d_in[28];

  float* out = (float*)d_out;
  float* o_reg   = out;              // 8*512*6
  float* o_risk  = out + 24576;      // 8*512*18
  float* o_warn  = out + 98304;      // 8*512*6
  float* o_tattn = out + 122880;     // 4096*48*48
  float* o_alpha = out + 9560064;    // 8*512*512

  char* ws = (char*)d_ws;
  size_t off = 0;
  auto take = [&](size_t bytes) { size_t o = off; off += (bytes + 255) & ~(size_t)255; return o; };
  float*    W0    = (float*)   (ws + take((size_t)MTOT * HH * 4));      // h / ln1 (f32)
  _Float16* W1    = (_Float16*)(ws + take((size_t)MTOT * HH * 2));      // h16 / ao16 / ln1_16
  _Float16* W2    = (_Float16*)(ws + take((size_t)MTOT * 3 * HH * 2));  // qkv16 / ffn1_16
  float*    W3    = (float*)   (ws + take((size_t)MTOT * HH * 4));      // gemm f32 / ln2
  _Float16* XR16  = (_Float16*)(ws + take((size_t)MTOT * 32 * 2));      // padded K=32
  _Float16* XS16  = (_Float16*)(ws + take((size_t)NN * 32 * 2));
  _Float16* CAT16 = (_Float16*)(ws + take((size_t)MG * 384 * 2));
  float*    GH32  = (float*)   (ws + take((size_t)MG * HH * 4));
  _Float16* GHT   = (_Float16*)(ws + take((size_t)MG * HH * 2));        // gh^T per batch
  _Float16* AL16  = (_Float16*)(ws + take((size_t)MG * NN * 2));
  float*    FUSED = (float*)   (ws + take((size_t)MG * HH * 4));
  float*    S1    = (float*)   (ws + take((size_t)MG * 4));
  float*    S2    = (float*)   (ws + take((size_t)MG * 4));
  // f16 weight copies (K-contiguous, padded where needed)
  _Float16* PW    = (_Float16*)(ws + take((size_t)HH * 32 * 2));        // proj_w 128x32
  _Float16* INW   = (_Float16*)(ws + take((size_t)384 * HH * 2));
  _Float16* OUTW  = (_Float16*)(ws + take((size_t)HH * HH * 2));
  _Float16* FW1   = (_Float16*)(ws + take((size_t)256 * HH * 2));
  _Float16* FW2   = (_Float16*)(ws + take((size_t)HH * 256 * 2));
  _Float16* STW   = (_Float16*)(ws + take((size_t)HH * 32 * 2));        // stat_w 128x32
  _Float16* GATW  = (_Float16*)(ws + take((size_t)HH * 256 * 2));
  _Float16* FUSEW = (_Float16*)(ws + take((size_t)HH * 384 * 2));

  // ---- Stage operands to f16 ------------------------------------------------
  xr_stage_kernel<<<(MTOT * 32 + 255) / 256, 256, 0, stream>>>(x, XR16);
  xs_stage_kernel<<<(NN * 32 + 255) / 256, 256, 0, stream>>>(x_static, XS16);
  stage_w(proj_w, PW,   HH, FF, 32, stream);
  stage_w(in_w,   INW,  384, HH, HH, stream);
  stage_w(out_w,  OUTW, HH, HH, HH, stream);
  stage_w(ffn_w1, FW1,  256, HH, HH, stream);
  stage_w(ffn_w2, FW2,  HH, 256, 256, stream);
  stage_w(stat_w, STW,  HH, SS, 32, stream);
  stage_w(gat_w,  GATW, HH, 256, 256, stream);
  stage_w(fuse_w, FUSEW, HH, 384, 384, stream);

  // ---- Temporal transformer -------------------------------------------------
  launch_gemm(XR16, 32, PW, 32, proj_b, W0, HH, W1, HH, MTOT, HH, 32, 0, stream);
  launch_gemm(W1, HH, INW, HH, in_b, nullptr, 0, W2, 3 * HH, MTOT, 3 * HH, HH, 0, stream);
  attn_kernel<<<MG, 128, 0, stream>>>(W2, W1, o_tattn);                 // ao16 -> W1
  launch_gemm(W1, HH, OUTW, HH, out_b, W3, HH, nullptr, 0, MTOT, HH, HH, 0, stream);
  add_ln_kernel<<<MTOT / 8, 256, 0, stream>>>(W0, W3, ln1_g, ln1_b, W0, W1, MTOT);
  launch_gemm(W1, HH, FW1, HH, ffn_b1, nullptr, 0, W2, 256, MTOT, 256, HH, 1, stream);
  launch_gemm(W2, 256, FW2, 256, ffn_b2, W3, HH, nullptr, 0, MTOT, HH, 256, 0, stream);
  add_ln_kernel<<<MTOT / 8, 256, 0, stream>>>(W0, W3, ln2_g, ln2_b, W3, nullptr, MTOT);

  // ---- GAT ------------------------------------------------------------------
  launch_gemm(XS16, 32, STW, 32, stat_b, FUSED /*srepr tmp*/, HH, nullptr, 0,
              NN, HH, 32, 2, stream);
  node_build_kernel<<<(MG * 256 + 255) / 256, 256, 0, stream>>>(W3, FUSED, CAT16);
  launch_gemm(CAT16, 384, GATW, 256, nullptr, GH32, HH, nullptr, 0, MG, HH, 256, 0, stream);
  ght_kernel<<<(MG * HH + 255) / 256, 256, 0, stream>>>(GH32, GHT);
  s12_kernel<<<MG / 8, 256, 0, stream>>>(GH32, gat_a, S1, S2);
  alpha_kernel<<<MG / 8, 256, 0, stream>>>(S1, S2, adj, o_alpha, AL16);
  for (int b = 0; b < BB; ++b)   // g_repr[b] = alpha[b] @ gh[b] via ghT (K=512)
    launch_gemm(AL16 + (size_t)b * NN * NN, NN,
                GHT + (size_t)b * HH * NN, NN, nullptr,
                nullptr, 0, CAT16 + (size_t)b * NN * 384 + 256, 384,
                NN, HH, NN, 0, stream);
  launch_gemm(CAT16, 384, FUSEW, 384, fuse_b, FUSED, HH, nullptr, 0, MG, HH, 384, 2, stream);
  heads_kernel<<<(MG * 30 + 255) / 256, 256, 0, stream>>>(
      FUSED, reg_w, reg_b, risk_w, risk_b, warn_w, warn_b, o_reg, o_risk, o_warn);
}
